// VGG16RoIHead_11836929868226
// MI455X (gfx1250) — matrix operand order repaired and checked by reading, not compiled, not run on hardware
//
#include <hip/hip_runtime.h>
#include <hip/hip_bf16.h>

// ---------------------------------------------------------------------------
// VGG16 RoI head for MI455X (gfx1250, wave32, WMMA + async global->LDS).
//   1) roi_pool    : x(2,512,37,50)+rois(128,5) -> A1 f16 (128 x 25088)
//   2) fc1 partial : A1 @ W1^T  (4-way K-split) -> P  f32 (4 x 128 x 4096)
//      reduce      : sum splits + b1 + ReLU     -> A2 f16 (128 x 4096)
//   3) fc2 partial : A2 @ W2^T  (4-way K-split) -> P
//      reduce      : sum splits + b2 + ReLU     -> A3 f16 (128 x 4096)
//   4) heads       : A3 @ W_loc^T / W_score^T   -> d_out f32 (128*84|128*21)
//
// FC GEMM (per wave: full M=128 x N=16 stripe over its K chunk):
//  - A slices (128 rows x 32 k, 8KB) staged to LDS with
//    GLOBAL_LOAD_ASYNC_TO_LDS_B128 (ASYNCcnt, no VGPR roundtrip), ping-pong
//    buffers, 80B padded row stride; single s_wait_asynccnt 0 before barrier,
//    issued *before* the WMMA block so staging latency hides behind math.
//  - Weights streamed fp32 once from HBM (nontemporal + prefetch) with a
//    2-step raw-register pipeline; cvt to f16 two steps after issue ->
//    v_wmma_f32_16x16x32_f16, f32 accumulate.
//  - 4-way K-split -> 1024 waves; deterministic fixed-order reduce.
// ---------------------------------------------------------------------------

typedef _Float16 v16h __attribute__((ext_vector_type(16)));
typedef _Float16 h8   __attribute__((ext_vector_type(8)));
typedef float    v8f  __attribute__((ext_vector_type(8)));
typedef float    v4f  __attribute__((ext_vector_type(4)));

#define KSPLIT 4
#define LDS_ROW 40   // _Float16 units: 32 data + 8 pad = 80B row stride
#define LDS_BUF_BYTES (128 * LDS_ROW * 2)   // 10240 B per buffer

// ------------------------------ ROI pooling --------------------------------
__global__ __launch_bounds__(256) void roipool_kernel(
    const float* __restrict__ x,     // (2,512,37,50)
    const float* __restrict__ rois,  // (128,5)
    _Float16* __restrict__ A1)       // (128,25088) f16
{
  constexpr int C = 512, H = 37, W = 50;
  const int n   = blockIdx.x;
  const int idx = blockIdx.y * blockDim.x + threadIdx.x;   // < 25088
  const int c   = idx / 49;
  const int rem = idx % 49;
  const int i   = rem / 7;
  const int j   = rem % 7;

  const float* r = rois + (size_t)n * 5;
  const int bidx = (int)r[0];
  const int sw = (int)floorf(r[1] * 0.0625f + 0.5f);
  const int sh = (int)floorf(r[2] * 0.0625f + 0.5f);
  const int ew = (int)floorf(r[3] * 0.0625f + 0.5f);
  const int eh = (int)floorf(r[4] * 0.0625f + 0.5f);
  const int rw = max(ew - sw + 1, 1);
  const int rh = max(eh - sh + 1, 1);
  const float bw = (float)rw / 7.0f;
  const float bh = (float)rh / 7.0f;

  int wlo = (int)floorf((float)j * bw) + sw;
  int whi = (int)ceilf(((float)j + 1.0f) * bw) + sw;
  int hlo = (int)floorf((float)i * bh) + sh;
  int hhi = (int)ceilf(((float)i + 1.0f) * bh) + sh;
  wlo = min(max(wlo, 0), W);  whi = min(max(whi, 0), W);
  hlo = min(max(hlo, 0), H);  hhi = min(max(hhi, 0), H);

  const float* feat = x + ((size_t)bidx * C + c) * (H * W);
  float m = -1.0e37f;
  for (int hh = hlo; hh < hhi; ++hh) {
    const float* row = feat + hh * W;
    for (int ww = wlo; ww < whi; ++ww)
      m = fmaxf(m, row[ww]);
  }
  if (m <= -5.0e36f) m = 0.0f;
  A1[(size_t)n * 25088 + idx] = (_Float16)m;
}

// ------------------------------ fragment helpers ---------------------------
struct BRaw { v4f x0, x1, x2, x3; };

// Issue 16 fp32 weight loads (k+8*kgrp+{0..7,16..23}) (nontemporal) -- no wait here.
__device__ inline BRaw load_b_raw(const float* __restrict__ wrow, int k,
                                  int kgrp) {
  const float* p = wrow + k + 8 * kgrp;
  BRaw r;
  r.x0 = __builtin_nontemporal_load((const v4f*)(p + 0));
  r.x1 = __builtin_nontemporal_load((const v4f*)(p + 4));
  r.x2 = __builtin_nontemporal_load((const v4f*)(p + 16));
  r.x3 = __builtin_nontemporal_load((const v4f*)(p + 20));
  return r;
}

// Convert raw fp32 B fragment to f16 WMMA layout (forces the load wait).
__device__ inline v16h cvt_b(const BRaw& r) {
  v16h d;
#pragma unroll
  for (int q = 0; q < 4; ++q) {
    d[q]      = (_Float16)r.x0[q];
    d[4 + q]  = (_Float16)r.x1[q];
    d[8 + q]  = (_Float16)r.x2[q];
    d[12 + q] = (_Float16)r.x3[q];
  }
  return d;
}

// A fragment (16x32 f16) from the LDS-staged slice.
// lane L: row=(L&15); half=L>>4 -> K-subranges {0..7,16..23}/{8..15,24..31}.
__device__ inline v16h lds_a_frag(const _Float16* lbuf, int row, int half) {
  const _Float16* p = lbuf + row * LDS_ROW + 8 * half;
  h8 lo = *(const h8*)p;         // K = 8*half   .. +7
  h8 hi = *(const h8*)(p + 16);  // K = 16+8*half.. +7
  v16h rdat;
#pragma unroll
  for (int q = 0; q < 8; ++q) { rdat[q] = lo[q]; rdat[8 + q] = hi[q]; }
  return rdat;
}

// A fragment straight from global f16 (used by the small head kernel).
__device__ inline v16h gbl_a_frag(const _Float16* __restrict__ A, int row,
                                  int K, int k, int half) {
  const _Float16* p = A + (size_t)row * K + k + 8 * half;
  h8 lo = *(const h8*)p;
  h8 hi = *(const h8*)(p + 16);
  v16h rdat;
#pragma unroll
  for (int q = 0; q < 8; ++q) { rdat[q] = lo[q]; rdat[8 + q] = hi[q]; }
  return rdat;
}

// Async copy of this thread's 64B row of the next A slice into LDS.
// IOFFSET is applied to BOTH the global and LDS address (ISA 08 §4.4).
__device__ inline void async_stage(unsigned lds_off,
                                   const _Float16* __restrict__ src) {
  asm volatile(
      "global_load_async_to_lds_b128 %0, %1, off\n\t"
      "global_load_async_to_lds_b128 %0, %1, off offset:16\n\t"
      "global_load_async_to_lds_b128 %0, %1, off offset:32\n\t"
      "global_load_async_to_lds_b128 %0, %1, off offset:48"
      :
      : "v"(lds_off), "v"(src)
      : "memory");
}

__device__ inline void wait_async0() {
  asm volatile("s_wait_asynccnt 0" ::: "memory");
}

// 8 WMMAs consuming one staged slice with a shared B fragment.
__device__ inline void consume_slice(const _Float16* lbuf, int lane, int kgrp,
                                     v16h b, v8f acc[8]) {
#pragma unroll
  for (int t = 0; t < 8; ++t) {
    v16h a = lds_a_frag(lbuf, t * 16 + (lane & 15), kgrp);
    acc[t] = __builtin_amdgcn_wmma_f32_16x16x32_f16(
        false, a, false, b, (short)0, acc[t], false, false);
  }
}

// ------------------------- FC GEMM partial (K-split) -----------------------
// Block: 128 threads = 4 waves, 4 consecutive N-tiles; blockIdx.y = K split.
// Each wave: M=128 x N=16 over chunk K/KSPLIT; fp32 partial out (no bias).
__global__ __launch_bounds__(128) void fc_gemm_partial_kernel(
    const _Float16* __restrict__ A,   // 128 x K f16
    const float*    __restrict__ Wt,  // N x K f32
    float*          __restrict__ P,   // KSPLIT x 128 x Nld f32 partials
    int K, int Nld)
{
  __shared__ _Float16 Asmem[2][128 * LDS_ROW];   // ping-pong, 2 x 10KB

  const int tid   = threadIdx.x;
  const int lane  = tid & 31;
  const int wave  = tid >> 5;
  const int ntile = blockIdx.x * 4 + wave;
  const int o     = ntile * 16 + (lane & 15);
  const int kgrp  = lane >> 4;
  const int split = blockIdx.y;
  const int chunk = K / KSPLIT;           // multiple of 64 for all our sizes
  const int k0    = split * chunk;
  const int kend  = k0 + chunk;
  const int nsteps = chunk / 32;          // even, >= 2

  const float*    wrow = Wt + (size_t)o * K;
  const _Float16* arow = A + (size_t)tid * K;    // staging row: row = tid

  const _Float16* sm0 = &Asmem[0][0];
  const _Float16* sm1 = &Asmem[1][0];
  const unsigned  lds0 = (unsigned)(unsigned long long)(sm0 + tid * LDS_ROW);
  const unsigned  lds1 = (unsigned)(unsigned long long)(sm1 + tid * LDS_ROW);

  v8f acc[8];
#pragma unroll
  for (int t = 0; t < 8; ++t)
#pragma unroll
    for (int q = 0; q < 8; ++q) acc[t][q] = 0.0f;

  // ---- prologue: async-stage slice 0 (even -> buf0); preload B steps 0,1 ---
  async_stage(lds0, arow + k0);
  BRaw raw0 = load_b_raw(wrow, k0, kgrp);        // step 0
  BRaw raw1 = load_b_raw(wrow, k0 + 32, kgrp);   // step 1
  wait_async0();
  __syncthreads();

  // ---- main loop, 2 steps per iteration (parities fixed: even->buf0) ------
  int s = 0;
  for (; s + 2 < nsteps; s += 2) {
    const int kk = k0 + s * 32;
    if (kk + 256 < kend)
      __builtin_prefetch((const void*)(wrow + kk + 256), 0, 1);

    // step s (even): stage s+1 -> buf1, consume buf0 with cvt(raw0),
    // issue raw B for step s+2.
    v16h bE = cvt_b(raw0);
    raw0 = load_b_raw(wrow, kk + 64, kgrp);
    async_stage(lds1, arow + kk + 32);
    consume_slice(sm0, lane, kgrp, bE, acc);
    wait_async0();
    __syncthreads();

    // step s+1 (odd): stage s+2 -> buf0, consume buf1 with cvt(raw1),
    // issue raw B for step s+3 (always valid inside this loop).
    v16h bO = cvt_b(raw1);
    raw1 = load_b_raw(wrow, kk + 96, kgrp);
    async_stage(lds0, arow + kk + 64);
    consume_slice(sm1, lane, kgrp, bO, acc);
    wait_async0();
    __syncthreads();
  }

  // ---- tail: steps nsteps-2 (even, buf0) and nsteps-1 (odd, buf1) ---------
  {
    const int kk = k0 + s * 32;
    async_stage(lds1, arow + kk + 32);           // stage last slice -> buf1
    consume_slice(sm0, lane, kgrp, cvt_b(raw0), acc);
    wait_async0();
    __syncthreads();
    consume_slice(sm1, lane, kgrp, cvt_b(raw1), acc);
  }

  // ---- store fp32 partials ----
  float* pout = P + ((size_t)split * 128) * Nld;
#pragma unroll
  for (int t = 0; t < 8; ++t) {
#pragma unroll
    for (int rr = 0; rr < 8; ++rr) {
      const int m = t * 16 + kgrp * 8 + rr;
      pout[(size_t)m * Nld + o] = acc[t][rr];
    }
  }
}

// --------------------- reduce splits + bias + ReLU -> f16 ------------------
__global__ __launch_bounds__(256) void reduce_bias_relu_kernel(
    const float* __restrict__ P, const float* __restrict__ bias,
    _Float16* __restrict__ outh, int Nld)
{
  const int idx = blockIdx.x * 256 + threadIdx.x;     // < 128*Nld
  const int o   = idx & (Nld - 1);                    // Nld = 4096
  float s = 0.0f;
#pragma unroll
  for (int sp = 0; sp < KSPLIT; ++sp)
    s += P[(size_t)sp * 128 * Nld + idx];
  s = fmaxf(s + bias[o], 0.0f);
  outh[idx] = (_Float16)s;
}

// ------------------------------ heads GEMM ---------------------------------
__global__ __launch_bounds__(32) void head_gemm_kernel(
    const _Float16* __restrict__ A,     // 128 x 4096 f16
    const float* __restrict__ Wloc, const float* __restrict__ bloc,
    const float* __restrict__ Wsc,  const float* __restrict__ bsc,
    float* __restrict__ out)            // [128*84 | 128*21]
{
  constexpr int K = 4096;
  const int lane = threadIdx.x & 31;
  const int o    = blockIdx.x * 16 + (lane & 15);
  const int kgrp = lane >> 4;

  const float* wrow;
  if (o < 84)       wrow = Wloc + (size_t)o * K;
  else if (o < 105) wrow = Wsc + (size_t)(o - 84) * K;
  else              wrow = Wsc;     // padding lanes: safe dummy row

  v8f acc[8];
#pragma unroll
  for (int t = 0; t < 8; ++t)
#pragma unroll
    for (int q = 0; q < 8; ++q) acc[t][q] = 0.0f;

  // register ping-pong over K (128 steps, even)
  v16h b0 = cvt_b(load_b_raw(wrow, 0, kgrp));
  v16h a0[8];
#pragma unroll
  for (int t = 0; t < 8; ++t)
    a0[t] = gbl_a_frag(A, t * 16 + (lane & 15), K, 0, kgrp);

  for (int k = 0; k < K; k += 64) {
    const int k1 = k + 32;                 // always < K (K multiple of 64)
    v16h b1 = cvt_b(load_b_raw(wrow, k1, kgrp));
    v16h a1[8];
#pragma unroll
    for (int t = 0; t < 8; ++t)
      a1[t] = gbl_a_frag(A, t * 16 + (lane & 15), K, k1, kgrp);
#pragma unroll
    for (int t = 0; t < 8; ++t)
      acc[t] = __builtin_amdgcn_wmma_f32_16x16x32_f16(
          false, a0[t], false, b0, (short)0, acc[t], false, false);
    const int k2 = k + 64;
    if (k2 < K) {
      b0 = cvt_b(load_b_raw(wrow, k2, kgrp));
#pragma unroll
      for (int t = 0; t < 8; ++t)
        a0[t] = gbl_a_frag(A, t * 16 + (lane & 15), K, k2, kgrp);
    }
#pragma unroll
    for (int t = 0; t < 8; ++t)
      acc[t] = __builtin_amdgcn_wmma_f32_16x16x32_f16(
          false, a1[t], false, b1, (short)0, acc[t], false, false);
  }

  const float bv = (o < 84) ? bloc[o] : ((o < 105) ? bsc[o - 84] : 0.0f);
#pragma unroll
  for (int t = 0; t < 8; ++t) {
#pragma unroll
    for (int rr = 0; rr < 8; ++rr) {
      const int m = t * 16 + kgrp * 8 + rr;
      const float v = acc[t][rr] + bv;
      if (o < 84)        out[(size_t)m * 84 + o] = v;
      else if (o < 105)  out[128 * 84 + (size_t)m * 21 + (o - 84)] = v;
    }
  }
}

// ------------------------------ launcher -----------------------------------
extern "C" void kernel_launch(void* const* d_in, const int* in_sizes, int n_in,
                              void* d_out, int out_size, void* d_ws, size_t ws_size,
                              hipStream_t stream) {
  const float* x      = (const float*)d_in[0];
  const float* rois   = (const float*)d_in[1];
  const float* W1     = (const float*)d_in[2];
  const float* b1     = (const float*)d_in[3];
  const float* W2     = (const float*)d_in[4];
  const float* b2     = (const float*)d_in[5];
  const float* W_loc  = (const float*)d_in[6];
  const float* b_loc  = (const float*)d_in[7];
  const float* W_sc   = (const float*)d_in[8];
  const float* b_sc   = (const float*)d_in[9];
  float* out = (float*)d_out;

  // workspace layout (16B aligned offsets)
  _Float16* A1 = (_Float16*)d_ws;                                    // 6,422,528 B
  _Float16* A2 = (_Float16*)((char*)d_ws + (size_t)128 * 25088 * 2); // 1,048,576 B
  _Float16* A3 = A2 + (size_t)128 * 4096;                            // 1,048,576 B
  float*    P  = (float*)((char*)d_ws + 8519680);                    // 8,388,608 B

  // 1) ROI max-pool -> f16 A-matrix (25088 = 98 * 256 per roi)
  roipool_kernel<<<dim3(128, 98), 256, 0, stream>>>(x, rois, A1);

  // 2) FC1: 256 N-tiles / 4 waves per block = 64 blocks x KSPLIT
  fc_gemm_partial_kernel<<<dim3(64, KSPLIT), 128, 0, stream>>>(A1, W1, P,
                                                               25088, 4096);
  reduce_bias_relu_kernel<<<2048, 256, 0, stream>>>(P, b1, A2, 4096);

  // 3) FC2
  fc_gemm_partial_kernel<<<dim3(64, KSPLIT), 128, 0, stream>>>(A2, W2, P,
                                                               4096, 4096);
  reduce_bias_relu_kernel<<<2048, 256, 0, stream>>>(P, b2, A3, 4096);

  // 4) heads: 105 cols -> 7 tiles of 16
  head_gemm_kernel<<<7, 32, 0, stream>>>(A3, W_loc, b_loc, W_sc, b_sc, out);
}
